// Attention_Emb_4922032521314
// MI455X (gfx1250) — matrix operand order repaired and checked by reading, not compiled, not run on hardware
//
#include <hip/hip_runtime.h>
#include <hip/hip_bf16.h>
#include <math.h>

// ---------------------------------------------------------------------------
// Attention_Emb on MI455X (gfx1250, wave32, WMMA).
// Compute-bound (~10 GFLOP over <20MB, L2-resident) -> all GEMM stages via
// v_wmma_f32_16x16x32_bf16. All intermediates stored pre-swizzled in WMMA
// fragment order so every fragment load is a contiguous 32B vector load and
// fragment-layout stores are packed b128 where runs are contiguous.
// Flash phase-3 V fragments are software-pipelined into LDS with
// global_load_async_to_lds_b128 / s_wait_asynccnt (CDNA5 async path).
// pos-attention branch is separable -> one softmax vector + p.V per (b,h).
// ---------------------------------------------------------------------------

#define B_    4
#define S_    1024
#define W_    8
#define DIM_  64
#define H_    4
#define HD_   128   // head dim = W*DIM/H
#define SW_   8192  // S_*W_
#define ROWS_ 32768 // B_*S_*W_

typedef __attribute__((ext_vector_type(16))) __bf16 v16bf;
typedef __attribute__((ext_vector_type(8)))  float  v8f;

static __device__ __forceinline__ __bf16 f2bf(float f) { return (__bf16)f; }
static __device__ __forceinline__ float  bf2f(__bf16 b) { return (float)b; }

union pk8 { __bf16 h[8]; uint4 u; };

// A-operand (16x32 bf16): element e of lane -> local K index
static __device__ __forceinline__ int a_klocal(int lane, int e) {
  return ((lane >> 4) << 4) + e;
}
// inverse: (row M, local k kl) -> (lane, elem) of the A fragment
static __device__ __forceinline__ int frag_lane(int M, int kl) {
  return M + ((kl >> 4) << 4);
}
static __device__ __forceinline__ int frag_elem(int kl) {
  return kl & 15;
}

// ---- CDNA5 async global->LDS copy (ISA 08_async_tensor) -------------------
// LDS byte address = low 32 bits of the flat shared pointer (ISA 10.2).
static __device__ __forceinline__ void async_ld_b128(void* lds, const void* g) {
  unsigned l = (unsigned)(unsigned long long)lds;
  asm volatile("global_load_async_to_lds_b128 %0, %1, off"
               :: "v"(l), "v"(g) : "memory");
}
static __device__ __forceinline__ void wait_async_le2() {
  asm volatile("s_wait_asynccnt 0x2" ::: "memory");
}
static __device__ __forceinline__ void wait_async_0() {
  asm volatile("s_wait_asynccnt 0x0" ::: "memory");
}

// ---------------- strength projection: s_vec[64] ---------------------------
__global__ void k_strength(const float* __restrict__ strength,
                           const float* __restrict__ str_w,
                           const float* __restrict__ str_b,
                           float* __restrict__ s_vec) {
  int d = threadIdx.x;
  if (d < DIM_) {
    float acc = str_b[d];
    const float4* wp = (const float4*)(str_w + d * 512);
    const float4* sp = (const float4*)strength;
    for (int c = 0; c < 128; ++c) {
      float4 wv = wp[c], sv = sp[c];
      acc += wv.x * sv.x + wv.y * sv.y + wv.z * sv.z + wv.w * sv.w;
    }
    s_vec[d] = acc;
  }
}

// -------- pos MLP + head dot: a_score[b,h,t] = ph[b,t] . head_w[h] ---------
__global__ void k_posmlp(const float* __restrict__ pos,
                         const float* __restrict__ w1, const float* __restrict__ b1,
                         const float* __restrict__ w2, const float* __restrict__ b2,
                         const float* __restrict__ head_w,
                         float* __restrict__ a_score) {
  int idx = blockIdx.x * blockDim.x + threadIdx.x;
  if (idx >= B_ * S_) return;
  int b = idx >> 10, t = idx & 1023;
  float p0 = pos[idx * 3 + 0], p1 = pos[idx * 3 + 1], p2 = pos[idx * 3 + 2];
  float hdn[3];
  #pragma unroll
  for (int i = 0; i < 3; ++i) {
    float v = w1[i * 3] * p0 + w1[i * 3 + 1] * p1 + w1[i * 3 + 2] * p2 + b1[i];
    hdn[i] = fmaxf(v, 0.f);
  }
  float ph[8];
  #pragma unroll
  for (int k = 0; k < 8; ++k)
    ph[k] = w2[k * 3] * hdn[0] + w2[k * 3 + 1] * hdn[1] + w2[k * 3 + 2] * hdn[2] + b2[k];
  #pragma unroll
  for (int h = 0; h < H_; ++h) {
    float a = 0.f;
    #pragma unroll
    for (int k = 0; k < 8; ++k) a += head_w[h * 8 + k] * ph[k];
    a_score[((size_t)b * H_ + h) * S_ + t] = a;
  }
}

// ------ pos softmax (s-independent): pvec[bh,t] = softmax_t(-a[t]) ---------
__global__ void k_possoftmax(const float* __restrict__ a_score,
                             float* __restrict__ pvec) {
  __shared__ float red[256];
  int bh = blockIdx.x, tid = threadIdx.x;
  const float* a = a_score + (size_t)bh * S_;
  float mx = -1e30f;
  for (int i = tid; i < S_; i += 256) mx = fmaxf(mx, -a[i]);
  red[tid] = mx; __syncthreads();
  for (int s2 = 128; s2 > 0; s2 >>= 1) {
    if (tid < s2) red[tid] = fmaxf(red[tid], red[tid + s2]);
    __syncthreads();
  }
  float m = red[0]; __syncthreads();
  float sum = 0.f;
  for (int i = tid; i < S_; i += 256) sum += __expf(-a[i] - m);
  red[tid] = sum; __syncthreads();
  for (int s2 = 128; s2 > 0; s2 >>= 1) {
    if (tid < s2) red[tid] += red[tid + s2];
    __syncthreads();
  }
  float inv = 1.f / red[0];
  for (int i = tid; i < S_; i += 256)
    pvec[(size_t)bh * S_ + i] = __expf(-a[i] - m) * inv;
}

// ---- QKV projection via WMMA: Y = xt@emb.T + s ----------------------------
// Tile rows = 16 consecutive s for fixed (b,w): h = w>>1, e = (w&1)*64+dcol
// are lane-constant, so every output store is a fixed-stride pointer walk.
// Layouts produced:
//   Qswz[(((bh*64+st)*4+c)<<9) + lane*16 + elem]   (flash A-fragments)
//   Kb  [bh,t,e] row-major                         (flash B-frags contiguous)
//   Vswz[(((bh*32+kc)*8+nt)<<9) + lane*16 + elem]  (flash phase-3 B-frags)
__global__ void k_proj(const float* __restrict__ x,
                       const float* __restrict__ qe_all,
                       const float* __restrict__ ke_all,
                       const float* __restrict__ ve_all,
                       const int* __restrict__ embed_id,
                       const float* __restrict__ s_vec,
                       __bf16* __restrict__ Qswz, __bf16* __restrict__ Kb,
                       __bf16* __restrict__ Vswz) {
  const int id = *embed_id;
  const float* embs[3] = { qe_all + (size_t)id * DIM_ * DIM_,
                           ke_all + (size_t)id * DIM_ * DIM_,
                           ve_all + (size_t)id * DIM_ * DIM_ };

  int lane = threadIdx.x & 31;
  int wv = threadIdx.x >> 5;
  int tile = blockIdx.x * 4 + wv;          // ((b*8 + w)*64 + st)
  int st = tile & 63;
  int bw = tile >> 6;
  int w = bw & 7, b = bw >> 3;
  int s0 = st * 16;
  int h = w >> 1;
  size_t bh = (size_t)b * H_ + h;
  int hi = lane >> 4;                      // half-wave
  int Mbase = hi << 3;

  // A fragments: rows s0..s0+15 of xt for this (b,w); K = channel c
  const float* xbase = x + (size_t)b * DIM_ * SW_ +
                       (size_t)(s0 + (lane & 15)) * W_ + w;
  v16bf a0, a1;
  #pragma unroll
  for (int e = 0; e < 16; ++e) {
    int kl = a_klocal(lane, e);
    a0[e] = f2bf(xbase[(size_t)kl * SW_]);
    a1[e] = f2bf(xbase[(size_t)(32 + kl) * SW_]);
  }

  #pragma unroll
  for (int e3 = 0; e3 < 3; ++e3) {
    const float* emb = embs[e3];
    #pragma unroll
    for (int nt = 0; nt < 4; ++nt) {
      int dcol = nt * 16 + (lane & 15);
      int kb2 = hi << 4;
      // B[k=c][n=d] = emb[d*64+c]; contiguous in c -> float4 loads + cvt
      const float4* ep0 = (const float4*)(emb + dcol * 64 + kb2);
      const float4* ep1 = (const float4*)(emb + dcol * 64 + 32 + kb2);
      v16bf b0, b1;
      #pragma unroll
      for (int q4 = 0; q4 < 4; ++q4) {
        float4 f0 = ep0[q4], f1 = ep1[q4];
        b0[q4 * 4 + 0] = f2bf(f0.x); b0[q4 * 4 + 1] = f2bf(f0.y);
        b0[q4 * 4 + 2] = f2bf(f0.z); b0[q4 * 4 + 3] = f2bf(f0.w);
        b1[q4 * 4 + 0] = f2bf(f1.x); b1[q4 * 4 + 1] = f2bf(f1.y);
        b1[q4 * 4 + 2] = f2bf(f1.z); b1[q4 * 4 + 3] = f2bf(f1.w);
      }
      v8f acc = {};
      acc = __builtin_amdgcn_wmma_f32_16x16x32_bf16(false, a0, false, b0,
                                                    (short)0, acc, false, false);
      acc = __builtin_amdgcn_wmma_f32_16x16x32_bf16(false, a1, false, b1,
                                                    (short)0, acc, false, false);
      float sadd = s_vec[dcol];
      int ecol = ((w & 1) << 6) + dcol;    // head-dim column (lane-constant)

      if (e3 == 0) {                       // Q: A-fragment swizzle
        int c = ecol >> 5, kl = ecol & 31;
        __bf16* qp = Qswz + (((bh * 64 + st) * 4 + c) << 9) +
                     frag_lane(Mbase, kl) * 16 + frag_elem(kl);
        #pragma unroll
        for (int rr = 0; rr < 8; ++rr)
          qp[rr * 16] = f2bf(acc[rr] + sadd);     // stride 32B per row
      } else if (e3 == 1) {                // K: row-major
        __bf16* kp = Kb + (bh * S_ + s0 + Mbase) * HD_ + ecol;
        #pragma unroll
        for (int rr = 0; rr < 8; ++rr)
          kp[rr * HD_] = f2bf(acc[rr] + sadd);    // stride 256B per row
      } else {                             // V: B-fragment swizzle
        int kc = st >> 1, nt2 = ecol >> 4;
        int lane2 = (ecol & 15) + ((st & 1) << 4);
        __bf16* vp = Vswz + (((bh * 32 + kc) * 8 + nt2) << 9) +
                     lane2 * 16 + Mbase;
        pk8 pk;                             // 8 contiguous bf16 -> one b128
        #pragma unroll
        for (int rr = 0; rr < 8; ++rr) pk.h[rr] = f2bf(acc[rr] + sadd);
        *(uint4*)vp = pk.u;
      }
    }
  }
}

// --------------- pv[bh,e] = sum_t pvec[t] * V[bh,t,e] ----------------------
// V swizzle makes (t&15) the fastest axis -> contiguous 16-element runs.
__global__ void k_pv(const float* __restrict__ pvec,
                     const __bf16* __restrict__ Vswz,
                     float* __restrict__ pv) {
  int bh = blockIdx.x, e = threadIdx.x;
  const float* p = pvec + (size_t)bh * S_;
  float acc = 0.f;
  for (int kc = 0; kc < 32; ++kc) {
    #pragma unroll
    for (int half = 0; half < 2; ++half) {
      const __bf16* vp = Vswz +
          ((((size_t)bh * 32 + kc) * 8 + (e >> 4)) << 9) +
          ((e & 15) + (half << 4)) * 16;
      v16bf vv = *(const v16bf*)vp;
      int t0 = kc * 32 + half * 16;
      #pragma unroll
      for (int tl = 0; tl < 16; ++tl) acc += p[t0 + tl] * bf2f(vv[tl]);
    }
  }
  pv[(size_t)bh * HD_ + e] = acc;
}

// ----- fused flash attention per (b,h,16-row s-tile); 8 waves/block --------
__global__ void k_flash(const __bf16* __restrict__ Qswz,
                        const __bf16* __restrict__ Kb,
                        const __bf16* __restrict__ Vswz,
                        const float* __restrict__ pv,
                        const float* __restrict__ gate,
                        __bf16* __restrict__ Yswz) {
  extern __shared__ char smem[];
  float*  sc     = (float*)smem;                             // [16][1024] f32
  __bf16* Ps     = (__bf16*)(smem + 65536);                   // 32KB, swizzled P
  float*  red    = (float*)(smem + 65536 + 32768);            // [16][16]
  float*  rowmax = red + 256;                                 // [16]
  float*  rowsum = rowmax + 16;                               // [16]
  char*   vbuf   = smem + 65536 + 32768 + 1152;               // 8 waves x 2KB

  int lane = threadIdx.x & 31;
  int wv   = threadIdx.x >> 5;                 // 0..7
  int st   = blockIdx.x;                       // s-tile
  int s0   = st * 16;
  int h    = blockIdx.y, b = blockIdx.z;
  size_t bh = (size_t)b * H_ + h;
  const __bf16* Kbase = Kb + bh * S_ * HD_;

  // Q fragments: contiguous vector loads from the pre-swizzled layout
  v16bf aq[4];
  #pragma unroll
  for (int c = 0; c < 4; ++c)
    aq[c] = *(const v16bf*)(Qswz + (((bh * 64 + st) * 4 + c) << 9) + lane * 16);

  const float scale = 0.0883883476483184f;     // 1/sqrt(128)

  // phase 1: score tile; wave wv owns t in [wv*128, wv*128+128)
  for (int tt = 0; tt < 8; ++tt) {
    int t0 = wv * 128 + tt * 16;
    v8f acc = {};
    #pragma unroll
    for (int c = 0; c < 4; ++c) {
      // B[k=e][n=t] = K[t,e]: 16 consecutive e per lane -> vector load
      v16bf bk = *(const v16bf*)(Kbase + (size_t)(t0 + (lane & 15)) * HD_ +
                                 c * 32 + ((lane >> 4) << 4));
      acc = __builtin_amdgcn_wmma_f32_16x16x32_bf16(false, aq[c], false, bk,
                                                    (short)0, acc, false, false);
    }
    #pragma unroll
    for (int rr = 0; rr < 8; ++rr) {
      int M = rr + ((lane >> 4) << 3);
      sc[M * 1024 + t0 + (lane & 15)] = acc[rr] * scale;
    }
  }
  __syncthreads();

  // phase 2: rowwise max / exp / sum ; P written in A-fragment swizzle.
  // Each 8 consecutive columns form one contiguous 16B run -> ds_store_b128.
  {
    int row = threadIdx.x >> 4;                // 0..15
    int seg = threadIdx.x & 15;
    float mx = -1e30f;
    for (int c = 0; c < 64; ++c) mx = fmaxf(mx, sc[row * 1024 + seg * 64 + c]);
    red[row * 16 + seg] = mx;
    __syncthreads();
    if (seg == 0) {
      float m2 = red[row * 16];
      for (int i = 1; i < 16; ++i) m2 = fmaxf(m2, red[row * 16 + i]);
      rowmax[row] = m2;
    }
    __syncthreads();
    float rm = rowmax[row];
    float sum = 0.f;
    #pragma unroll
    for (int cc = 0; cc < 2; ++cc) {           // two 32-col chunks per seg
      int kc = seg * 2 + cc;
      #pragma unroll
      for (int run = 0; run < 4; ++run) {      // kl = b4*16 + b3*8 + i
        int b3 = run & 1, b4 = run >> 1;
        pk8 pk;
        #pragma unroll
        for (int i = 0; i < 8; ++i) {
          int col = kc * 32 + b4 * 16 + b3 * 8 + i;
          float ev = __expf(sc[row * 1024 + col] - rm);
          sum += ev;
          pk.h[i] = f2bf(ev);
        }
        *(uint4*)(Ps + kc * 512 + (row + (b4 << 4)) * 16 + (b3 << 3)) = pk.u;
      }
    }
    red[row * 16 + seg] = sum;
    __syncthreads();
    if (seg == 0) {
      float s2 = 0.f;
      for (int i = 0; i < 16; ++i) s2 += red[row * 16 + i];
      rowsum[row] = s2;
    }
    __syncthreads();
  }

  // phase 3: P @ V ; wave wv owns head-dim columns [wv*16, wv*16+16).
  // V fragments double-buffered via async global->LDS copies.
  char* vb = vbuf + wv * 2048;
  auto issueV = [&](int kc, int buf) {
    const __bf16* src = Vswz + (((bh * 32 + kc) * 8 + wv) << 9) + lane * 16;
    char* dst = vb + buf * 1024 + lane * 32;
    async_ld_b128(dst, src);
    async_ld_b128(dst + 16, src + 8);
  };
  v8f acc = {};
  issueV(0, 0);
  for (int kc = 0; kc < 32; ++kc) {
    if (kc + 1 < 32) { issueV(kc + 1, (kc + 1) & 1); wait_async_le2(); }
    else             { wait_async_0(); }
    v16bf ap = *(const v16bf*)(Ps + kc * 512 + lane * 16);
    v16bf bv = *(const v16bf*)(vb + (kc & 1) * 1024 + lane * 32);
    acc = __builtin_amdgcn_wmma_f32_16x16x32_bf16(false, ap, false, bv,
                                                  (short)0, acc, false, false);
  }

  // epilogue: (1-g)*softmax@V + g*(p.V); weights already sum to 1.
  // Store pre-swizzled for the out-projection A-fragments:
  //   idx = base0 + (s&~1)<<9 + (s&1)<<7  (derived affine form)
  float g = 1.f / (1.f + __expf(-gate[h]));
  int e = wv * 16 + (lane & 15);
  int wq = h * 2 + (e >> 6);                   // source-view index
  int d = e & 63;
  int c2 = d >> 5, kl = d & 31;
  float pvv = pv[bh * HD_ + e];
  size_t base0 = (((size_t)b * 1024 + c2) << 9) +
                 frag_lane(wq, kl) * 16 + frag_elem(kl);
  #pragma unroll
  for (int rr = 0; rr < 8; ++rr) {
    int M = rr + ((lane >> 4) << 3);
    int s = s0 + M;
    float val = (1.f - g) * (acc[rr] / rowsum[M]) + g * pvv;
    Yswz[base0 + ((size_t)(s & ~1) << 9) + ((s & 1) << 7)] = f2bf(val);
  }
}

// ------------- final projection: out = Y @ out_w.T + out_b -----------------
__global__ void k_outproj(const __bf16* __restrict__ Yswz,
                          const float* __restrict__ out_w,
                          const float* __restrict__ out_b,
                          float* __restrict__ out) {
  int lane = threadIdx.x & 31, wv = threadIdx.x >> 5;
  int tile = blockIdx.x * 4 + wv;              // 16-row tile
  int row_base = tile * 16;
  int Mbase = (lane >> 4) << 3;
  v16bf a0 = *(const v16bf*)(Yswz + ((tile * 2 + 0) << 9) + lane * 16);
  v16bf a1 = *(const v16bf*)(Yswz + ((tile * 2 + 1) << 9) + lane * 16);
  #pragma unroll
  for (int nt = 0; nt < 4; ++nt) {
    int d = nt * 16 + (lane & 15);
    int kb2 = (lane >> 4) << 4;
    const float4* ep0 = (const float4*)(out_w + d * 64 + kb2);
    const float4* ep1 = (const float4*)(out_w + d * 64 + 32 + kb2);
    v16bf b0, b1;
    #pragma unroll
    for (int q4 = 0; q4 < 4; ++q4) {
      float4 f0 = ep0[q4], f1 = ep1[q4];
      b0[q4 * 4 + 0] = f2bf(f0.x); b0[q4 * 4 + 1] = f2bf(f0.y);
      b0[q4 * 4 + 2] = f2bf(f0.z); b0[q4 * 4 + 3] = f2bf(f0.w);
      b1[q4 * 4 + 0] = f2bf(f1.x); b1[q4 * 4 + 1] = f2bf(f1.y);
      b1[q4 * 4 + 2] = f2bf(f1.z); b1[q4 * 4 + 3] = f2bf(f1.w);
    }
    v8f acc = {};
    acc = __builtin_amdgcn_wmma_f32_16x16x32_bf16(false, a0, false, b0,
                                                  (short)0, acc, false, false);
    acc = __builtin_amdgcn_wmma_f32_16x16x32_bf16(false, a1, false, b1,
                                                  (short)0, acc, false, false);
    float bias = out_b[d];
    float* op = out + (size_t)(row_base + Mbase) * DIM_ + d;
    #pragma unroll
    for (int rr = 0; rr < 8; ++rr)
      op[rr * DIM_] = acc[rr] + bias;           // stride 256B per row
  }
}

// ---------------------------------------------------------------------------
extern "C" void kernel_launch(void* const* d_in, const int* in_sizes, int n_in,
                              void* d_out, int out_size, void* d_ws, size_t ws_size,
                              hipStream_t stream) {
  (void)in_sizes; (void)n_in; (void)out_size; (void)ws_size;
  const float* x        = (const float*)d_in[0];
  const float* pos      = (const float*)d_in[1];
  const float* strength = (const float*)d_in[2];
  const int*   embed_id = (const int*)d_in[3];
  const float* q_emb    = (const float*)d_in[4];
  const float* k_emb    = (const float*)d_in[5];
  const float* v_emb    = (const float*)d_in[6];
  const float* pos_w1   = (const float*)d_in[7];
  const float* pos_b1   = (const float*)d_in[8];
  const float* pos_w2   = (const float*)d_in[9];
  const float* pos_b2   = (const float*)d_in[10];
  const float* head_w   = (const float*)d_in[11];
  // d_in[12] = head_b: cancels inside the pos softmax (constant per row).
  const float* gate     = (const float*)d_in[13];
  const float* out_w    = (const float*)d_in[14];
  const float* out_b    = (const float*)d_in[15];
  const float* str_w    = (const float*)d_in[16];
  const float* str_b    = (const float*)d_in[17];
  float* out = (float*)d_out;

  char* wp = (char*)d_ws;
  auto alloc = [&](size_t bytes) {
    char* p = wp; wp += (bytes + 255) & ~(size_t)255; return p;
  };
  float*  s_vec   = (float*)alloc(DIM_ * 4);
  float*  a_score = (float*)alloc((size_t)B_ * H_ * S_ * 4);
  float*  pvec    = (float*)alloc((size_t)B_ * H_ * S_ * 4);
  float*  pv      = (float*)alloc((size_t)B_ * H_ * HD_ * 4);
  __bf16* Qswz    = (__bf16*)alloc((size_t)B_ * H_ * S_ * HD_ * 2);
  __bf16* Kb      = (__bf16*)alloc((size_t)B_ * H_ * S_ * HD_ * 2);
  __bf16* Vswz    = (__bf16*)alloc((size_t)B_ * H_ * S_ * HD_ * 2);
  __bf16* Yswz    = (__bf16*)alloc((size_t)ROWS_ * DIM_ * 2);

  k_strength<<<1, 64, 0, stream>>>(strength, str_w, str_b, s_vec);
  k_posmlp<<<(B_ * S_ + 255) / 256, 256, 0, stream>>>(
      pos, pos_w1, pos_b1, pos_w2, pos_b2, head_w, a_score);
  k_possoftmax<<<B_ * H_, 256, 0, stream>>>(a_score, pvec);
  k_proj<<<ROWS_ / 64, 128, 0, stream>>>(
      x, q_emb, k_emb, v_emb, embed_id, s_vec, Qswz, Kb, Vswz);
  k_pv<<<B_ * H_, HD_, 0, stream>>>(pvec, Vswz, pv);
  size_t flash_lds = 65536 + 32768 + 1152 + 8 * 2048;   // sc + Ps + red + vbuf
  k_flash<<<dim3(S_ / 16, H_, B_), 256, flash_lds, stream>>>(
      Qswz, Kb, Vswz, pv, gate, Yswz);
  k_outproj<<<ROWS_ / 64, 128, 0, stream>>>(Yswz, out_w, out_b, out);
}
